// Head_7911329759304
// MI455X (gfx1250) — matrix operand order repaired and checked by reading, compiled, not run on hardware
//
#include <hip/hip_runtime.h>

typedef __attribute__((ext_vector_type(16))) __bf16 v16bf;
typedef __attribute__((ext_vector_type(8)))  __bf16 v8bf;
typedef __attribute__((ext_vector_type(8)))  float  v8f;
typedef __attribute__((ext_vector_type(4)))  float  v4f;

#define B_  512
#define T_  256
#define C_  384
#define HS_ 64

__device__ __forceinline__ v16bf cat16(v8bf lo, v8bf hi) {
  return __builtin_shufflevector(lo, hi, 0,1,2,3,4,5,6,7,8,9,10,11,12,13,14,15);
}

// ---------------------------------------------------------------------------
// Kernel 0: convert Wk/Wq/Wv (f32 [HS,C]) to bf16 in workspace; fold the
// attention scale C^-0.5 into Wq so kernel 2 needs no extra multiply.
// Order in workspace: [Wk | Wq | Wv], each HS_*C_ elements.
// ---------------------------------------------------------------------------
__global__ void wconv_kernel(const float* __restrict__ Wk,
                             const float* __restrict__ Wq,
                             const float* __restrict__ Wv,
                             __bf16* __restrict__ wbf, float qscale) {
  int i = blockIdx.x * blockDim.x + threadIdx.x;
  const int WN = HS_ * C_;
  if (i >= 3 * WN) return;
  int mat = i / WN, idx = i - mat * WN;
  float v;
  if (mat == 0)      v = Wk[idx];
  else if (mat == 1) v = Wq[idx] * qscale;
  else               v = Wv[idx];
  wbf[i] = (__bf16)v;
}

// ---------------------------------------------------------------------------
// Kernel 1: fused QKV projection.
// Each wave computes 16 token-rows x 64 cols for all three outputs.
// A = x rows (f32 -> bf16 on the fly), B = bf16 weights (row n of W is the
// K-major column of B, contiguous loads).
// Outputs: k,q as [B*T, HS] bf16; v transposed as [B, HS, T] bf16.
// ---------------------------------------------------------------------------
__global__ __launch_bounds__(256) void qkv_proj_kernel(
    const float* __restrict__ x, const __bf16* __restrict__ wbf,
    __bf16* __restrict__ kbuf, __bf16* __restrict__ qbuf,
    __bf16* __restrict__ vtbuf) {
  const int lane  = threadIdx.x & 31;
  const int wid   = threadIdx.x >> 5;
  const int mbase = (blockIdx.x * 8 + wid) * 16;
  const int lrow  = lane & 15;
  const int hi    = (lane < 16) ? 0 : 1;
  const int c0    = hi ? 8 : 0;    // A-fragment chunk offset
  const int koff  = hi ? 16 : 0;   // unused; B loads also use c0

  v8f acc[3][4];
#pragma unroll
  for (int m = 0; m < 3; ++m)
#pragma unroll
    for (int n = 0; n < 4; ++n)
#pragma unroll
      for (int r = 0; r < 8; ++r) acc[m][n][r] = 0.0f;

  const float* xr = x + (size_t)(mbase + lrow) * C_;

  for (int ks = 0; ks < C_ / 32; ++ks) {
    const int kc = ks * 32;
    // A fragment: 16x32 bf16, converted from f32
    v16bf af;
    {
      const float* p0 = xr + kc + c0;
      const float* p1 = xr + kc + 16 + c0;
      v4f a0 = *(const v4f*)p0;
      v4f a1 = *(const v4f*)(p0 + 4);
      v4f a2 = *(const v4f*)p1;
      v4f a3 = *(const v4f*)(p1 + 4);
#pragma unroll
      for (int i = 0; i < 4; ++i) {
        af[i]      = (__bf16)a0[i];
        af[4 + i]  = (__bf16)a1[i];
        af[8 + i]  = (__bf16)a2[i];
        af[12 + i] = (__bf16)a3[i];
      }
    }
#pragma unroll
    for (int m = 0; m < 3; ++m) {
      const __bf16* wmat = wbf + (size_t)m * HS_ * C_;
#pragma unroll
      for (int n = 0; n < 4; ++n) {
        const __bf16* wr = wmat + (size_t)(n * 16 + lrow) * C_ + kc + c0;
        v8bf b0 = *(const v8bf*)wr;
        v8bf b1 = *(const v8bf*)(wr + 16);
        v16bf bf = cat16(b0, b1);
        acc[m][n] = __builtin_amdgcn_wmma_f32_16x16x32_bf16(
            false, af, false, bf, (short)0, acc[m][n], false, false);
      }
    }
  }

  // store: C-layout element (VGPR r, lane) -> row mbase+r+8*hi, col n*16+lrow
#pragma unroll
  for (int n = 0; n < 4; ++n) {
#pragma unroll
    for (int r = 0; r < 8; ++r) {
      const int m = mbase + r + hi * 8;
      const int h = n * 16 + lrow;
      kbuf[(size_t)m * HS_ + h] = (__bf16)acc[0][n][r];
      qbuf[(size_t)m * HS_ + h] = (__bf16)acc[1][n][r];
      const int bb = m >> 8;       // m / T_
      const int tt = m & (T_ - 1); // m % T_
      vtbuf[((size_t)bb * HS_ + h) * T_ + tt] = (__bf16)acc[2][n][r];
    }
  }
}

// ---------------------------------------------------------------------------
// Kernel 2: causal flash attention. One block = one batch b and 128 q-rows;
// each wave owns 16 q-rows, iterates s in pairs of 16-wide tiles (K=32 for
// the PV wmma). Online softmax with per-16-lane-group shuffles.
// ---------------------------------------------------------------------------
__global__ __launch_bounds__(256) void attn_kernel(
    const __bf16* __restrict__ qbuf, const __bf16* __restrict__ kbuf,
    const __bf16* __restrict__ vtbuf, float* __restrict__ out) {
  __shared__ __bf16 pls[8][16 * 32];   // per-wave P staging (A-layout reread)

  const int lane = threadIdx.x & 31;
  const int wid  = threadIdx.x >> 5;
  const int b    = blockIdx.x >> 1;
  const int r0   = (blockIdx.x & 1) * 128 + wid * 16;
  const int lrow = lane & 15;
  const int hi   = (lane < 16) ? 0 : 1;
  const int c0   = hi ? 8 : 0;
  const int koff = hi ? 16 : 0;

  // preload Q fragments (16 rows x 64, 2 K-steps)
  v16bf qf[2];
  {
    const __bf16* qr = qbuf + ((size_t)b * T_ + r0 + lrow) * HS_;
#pragma unroll
    for (int kk = 0; kk < 2; ++kk) {
      v8bf lo = *(const v8bf*)(qr + kk * 32 + c0);
      v8bf up = *(const v8bf*)(qr + kk * 32 + 16 + c0);
      qf[kk] = cat16(lo, up);
    }
  }

  v8f acc[4];
#pragma unroll
  for (int n = 0; n < 4; ++n)
#pragma unroll
    for (int r = 0; r < 8; ++r) acc[n][r] = 0.0f;

  float mrow[8], lsum[8];
#pragma unroll
  for (int r = 0; r < 8; ++r) { mrow[r] = -1e30f; lsum[r] = 0.0f; }

  const int nsp = (r0 + 16 + 31) >> 5;   // s-pairs needed (causal bound)

  for (int sp = 0; sp < nsp; ++sp) {
    const int sb = sp * 32;

    // ---- S = q @ k^T (scale already folded into q) ----
    v8f S0, S1;
#pragma unroll
    for (int r = 0; r < 8; ++r) { S0[r] = 0.0f; S1[r] = 0.0f; }
    {
      const __bf16* kr0 = kbuf + ((size_t)b * T_ + sb + lrow) * HS_;
      const __bf16* kr1 = kr0 + 16 * HS_;
#pragma unroll
      for (int kk = 0; kk < 2; ++kk) {
        v8bf a0 = *(const v8bf*)(kr0 + kk * 32 + c0);
        v8bf a1 = *(const v8bf*)(kr0 + kk * 32 + c0 + 16);
        S0 = __builtin_amdgcn_wmma_f32_16x16x32_bf16(
            false, qf[kk], false, cat16(a0, a1), (short)0, S0, false, false);
        v8bf b0 = *(const v8bf*)(kr1 + kk * 32 + c0);
        v8bf b1 = *(const v8bf*)(kr1 + kk * 32 + c0 + 16);
        S1 = __builtin_amdgcn_wmma_f32_16x16x32_bf16(
            false, qf[kk], false, cat16(b0, b1), (short)0, S1, false, false);
      }
    }

    // ---- causal mask: only the last (diagonal) pair can cross ----
    if (sp == nsp - 1) {
#pragma unroll
      for (int r = 0; r < 8; ++r) {
        const int rowm = r0 + r + hi * 8;
        const int s0   = sb + lrow;
        S0[r] = (s0 <= rowm)      ? S0[r] : -1e30f;
        S1[r] = (s0 + 16 <= rowm) ? S1[r] : -1e30f;
      }
    }

    // ---- online softmax (row = VGPR slot, spread over 16-lane group) ----
    float alpha[8];
#pragma unroll
    for (int r = 0; r < 8; ++r) {
      float t = fmaxf(S0[r], S1[r]);
      t = fmaxf(t, __shfl_xor(t, 1));
      t = fmaxf(t, __shfl_xor(t, 2));
      t = fmaxf(t, __shfl_xor(t, 4));
      t = fmaxf(t, __shfl_xor(t, 8));
      const float mn = fmaxf(mrow[r], t);
      alpha[r] = __expf(mrow[r] - mn);
      mrow[r]  = mn;
      const float p0 = __expf(S0[r] - mn);
      const float p1 = __expf(S1[r] - mn);
      S0[r] = p0; S1[r] = p1;
      float rs = p0 + p1;
      rs += __shfl_xor(rs, 1);
      rs += __shfl_xor(rs, 2);
      rs += __shfl_xor(rs, 4);
      rs += __shfl_xor(rs, 8);
      lsum[r] = lsum[r] * alpha[r] + rs;
    }
#pragma unroll
    for (int n = 0; n < 4; ++n)
#pragma unroll
      for (int r = 0; r < 8; ++r) acc[n][r] *= alpha[r];

    // ---- P (C-layout) -> LDS -> A-layout fragment ----
    __bf16* pw = &pls[wid][0];
#pragma unroll
    for (int r = 0; r < 8; ++r) {
      const int rr = r + hi * 8;
      pw[rr * 32 + lrow]      = (__bf16)S0[r];
      pw[rr * 32 + 16 + lrow] = (__bf16)S1[r];
    }
    v8bf plo = *(const v8bf*)(pw + lrow * 32 + c0);
    v8bf phi = *(const v8bf*)(pw + lrow * 32 + 16 + c0);
    v16bf pf = cat16(plo, phi);

    // ---- acc += P @ V  (V transposed: contiguous along s) ----
#pragma unroll
    for (int n = 0; n < 4; ++n) {
      const __bf16* vr =
          vtbuf + ((size_t)b * HS_ + n * 16 + lrow) * T_ + sb + c0;
      v8bf v0 = *(const v8bf*)vr;
      v8bf v1 = *(const v8bf*)(vr + 16);
      acc[n] = __builtin_amdgcn_wmma_f32_16x16x32_bf16(
          false, pf, false, cat16(v0, v1), (short)0, acc[n], false, false);
    }
  }

  // ---- epilogue: out = acc / l ----
#pragma unroll
  for (int r = 0; r < 8; ++r) {
    const float inv = 1.0f / lsum[r];
    const int m = r0 + r + hi * 8;
    float* op = out + ((size_t)b * T_ + m) * HS_;
#pragma unroll
    for (int n = 0; n < 4; ++n) op[n * 16 + lrow] = acc[n][r] * inv;
  }
}

// ---------------------------------------------------------------------------
extern "C" void kernel_launch(void* const* d_in, const int* in_sizes, int n_in,
                              void* d_out, int out_size, void* d_ws,
                              size_t ws_size, hipStream_t stream) {
  (void)in_sizes; (void)n_in; (void)out_size; (void)ws_size;
  const float* x  = (const float*)d_in[0];
  const float* Wk = (const float*)d_in[1];
  const float* Wq = (const float*)d_in[2];
  const float* Wv = (const float*)d_in[3];
  float* out = (float*)d_out;

  const size_t WN   = (size_t)3 * HS_ * C_;     // 73728 bf16
  const size_t QKV  = (size_t)B_ * T_ * HS_;    // 8388608 bf16 each
  char* ws = (char*)d_ws;
  __bf16* wbf   = (__bf16*)ws;
  __bf16* kbuf  = (__bf16*)(ws + WN * 2);
  __bf16* qbuf  = (__bf16*)(ws + (WN + QKV) * 2);
  __bf16* vtbuf = (__bf16*)(ws + (WN + 2 * QKV) * 2);

  const float qscale = 0.05103103630798287f;    // 384^-0.5

  wconv_kernel<<<(int)((3 * HS_ * C_ + 255) / 256), 256, 0, stream>>>(
      Wk, Wq, Wv, wbf, qscale);
  qkv_proj_kernel<<<(B_ * T_) / 128, 256, 0, stream>>>(
      x, wbf, kbuf, qbuf, vtbuf);
  attn_kernel<<<B_ * (T_ / 128), 256, 0, stream>>>(qbuf, kbuf, vtbuf, out);
}
